// VolatilityAwareDualPath_76416058130990
// MI455X (gfx1250) — matrix operand order repaired and checked by reading, not compiled, not run on hardware
//
#include <hip/hip_runtime.h>

// ---------------- types / helpers ----------------
typedef __attribute__((ext_vector_type(16))) __bf16   bf16x16;
typedef __attribute__((ext_vector_type(8)))  float    f32x8;
typedef __attribute__((ext_vector_type(4)))  float    f32x4;
typedef __attribute__((ext_vector_type(4)))  unsigned u32x4;
typedef __attribute__((ext_vector_type(2)))  unsigned u32x2;
typedef __attribute__((ext_vector_type(8)))  int      i32x8;
typedef __attribute__((ext_vector_type(4)))  int      i32x4;

union Frag16 { bf16x16 v; u32x4 q[2]; unsigned short h[16]; };
static_assert(sizeof(Frag16) == 32, "frag size");

#define WMMA_BF16(A, B, C) \
  __builtin_amdgcn_wmma_f32_16x16x32_bf16(false, (A), false, (B), (short)0, (C), false, false)

#if defined(__has_builtin)
#  if __has_builtin(__builtin_amdgcn_tensor_load_to_lds) && __has_builtin(__builtin_amdgcn_s_wait_tensorcnt)
#    define HAVE_TDM 1
#  endif
#endif
#ifndef HAVE_TDM
#  define HAVE_TDM 0
#endif

__device__ inline unsigned short f2bf(float f) {
  unsigned u = __float_as_uint(f);
  unsigned r = u + 0x7FFFu + ((u >> 16) & 1u);   // round-to-nearest-even
  return (unsigned short)(r >> 16);
}
__device__ inline float bf2f(unsigned short h) { return __uint_as_float(((unsigned)h) << 16); }
__device__ inline float sigm(float x)  { return 1.f / (1.f + __expf(-x)); }
__device__ inline float tanhf_(float x){ float e = __expf(2.f * x); return 1.f - 2.f / (e + 1.f); }

#define TT 512
#define BB 256
#define DD 128
#define GG 832   // 192 (gru_f) + 320 (hb_f) + 320 (hb_b)

#if HAVE_TDM
// Tensor Data Mover: 1-D DMA of `ndw` dwords from global into LDS (ISA ch.8 D# layout).
// D# group0: count=1 (valid), lds_addr, global_addr[56:0], type=2 ("image").
// D# group1: data_size=4B, tensor_dim0 = tile_dim0 = ndw, tensor_dim1 = tile_dim1 = 1.
__device__ inline void tdm_load_to_lds(const void* gsrc, unsigned lds_off, unsigned ndw) {
  unsigned long long ga = (unsigned long long)(uintptr_t)gsrc;
  u32x4 g0;
  g0[0] = 1u;                                               // count=1, is_restore=0, gather off
  g0[1] = lds_off;                                          // lds_addr (bytes)
  g0[2] = (unsigned)(ga & 0xFFFFFFFFu);                     // global_addr[31:0]
  g0[3] = (unsigned)((ga >> 32) & 0x01FFFFFFu) | 0x80000000u; // global_addr[56:32] | type=2
  i32x8 g1;
  g1[0] = 0x00020000;                                       // workgroup_mask=0, data_size=4B
  g1[1] = (int)((ndw & 0xFFFFu) << 16);                     // tensor_dim0[15:0] @bits63:48
  g1[2] = (int)(((ndw >> 16) & 0xFFFFu) | (1u << 16));      // tensor_dim0[31:16], tensor_dim1=1
  g1[3] = (int)(ndw << 16);                                 // tile_dim0 = ndw @bits127:112
  g1[4] = 1;                                                // tile_dim1=1, tile_dim2=0
  g1[5] = (int)ndw;                                         // tensor_dim0_stride[31:0]
  g1[6] = 0;
  g1[7] = 0;
  i32x4 z = {0, 0, 0, 0};
#if __clang_major__ >= 23
  i32x8 z8 = {0, 0, 0, 0, 0, 0, 0, 0};
  __builtin_amdgcn_tensor_load_to_lds(g0, g1, z, z, z8, 0); // 6-arg form (therock-10.0)
#else
  __builtin_amdgcn_tensor_load_to_lds(g0, g1, z, z, 0);     // 5-arg form (ROCm 7.2)
#endif
}
#endif

// ---------------- kernel 1: pack / transpose / bf16-convert weights ----------------
__global__ __launch_bounds__(256) void pack_kernel(
    const float* __restrict__ gf_Wi, const float* __restrict__ gf_bi,
    const float* __restrict__ gf_Wh,
    const float* __restrict__ hf_W,  const float* __restrict__ hf_b,
    const float* __restrict__ hf_U,  const float* __restrict__ hf_c,
    const float* __restrict__ hb_W,  const float* __restrict__ hb_b,
    const float* __restrict__ hb_U,  const float* __restrict__ hb_c,
    const float* __restrict__ attnp_W,
    unsigned short* __restrict__ WcatT, float* __restrict__ Bcat,
    unsigned short* __restrict__ WhB,
    unsigned short* __restrict__ UTf, unsigned short* __restrict__ UTb,
    unsigned short* __restrict__ APT)
{
  const int id = blockIdx.x * 256 + threadIdx.x;    // grid covers 832*128
  if (id < GG * DD) {                               // WcatT[g][d], column-major weights for B-frags
    int g = id >> 7, d = id & 127;
    float v = (g < 192) ? gf_Wi[g * 128 + d]
            : (g < 512) ? hf_W[d * 320 + (g - 192)]
                        : hb_W[d * 320 + (g - 512)];
    WcatT[id] = f2bf(v);
  }
  if (id < GG) {                                    // fused input biases
    Bcat[id] = (id < 192) ? gf_bi[id]
             : (id < 512) ? hf_b[id - 192] + hf_c[id - 192]
                          : hb_b[id - 512] + hb_c[id - 512];
  }
  if (id < 192 * 64) WhB[id] = f2bf(gf_Wh[id]);     // Wh row-major == B-layout for h@Wh^T
  if (id < 320 * 64) {                              // UT[n][k] = U[k][n]
    int g = id >> 6, k_ = id & 63;
    UTf[id] = f2bf(hf_U[k_ * 320 + g]);
    UTb[id] = f2bf(hb_U[k_ * 320 + g]);
  }
  if (id < 64 * 128) {                              // APT[n][k] = attnp_W[k][n]
    int g = id >> 7, k_ = id & 127;
    APT[id] = f2bf(attnp_W[k_ * 64 + g]);
  }
}

// ---------------- kernel 2: x -> bf16 ----------------
__global__ __launch_bounds__(256) void cvt_x_kernel(const float* __restrict__ x,
                                                    unsigned short* __restrict__ xbf)
{
  size_t i = ((size_t)blockIdx.x * 256 + threadIdx.x) * 4;
  f32x4 v = *(const f32x4*)(x + i);
  u32x2 o;
  o[0] = (unsigned)f2bf(v[0]) | ((unsigned)f2bf(v[1]) << 16);
  o[1] = (unsigned)f2bf(v[2]) | ((unsigned)f2bf(v[3]) << 16);
  *(u32x2*)(xbf + i) = o;
}

// ---------------- kernel 3: big WMMA GEMM: XG[t*B+b][g] = x(b,t,:)@Wcat + Bcat ----------------
__global__ __launch_bounds__(128) void gemm_xg_kernel(
    const unsigned short* __restrict__ xbf,
    const unsigned short* __restrict__ WcatT,
    const float* __restrict__ Bcat,
    unsigned short* __restrict__ XG)
{
  const int wave = threadIdx.x >> 5, lane = threadIdx.x & 31;
  const int half = lane >> 4, n = lane & 15;
  const int mbase = blockIdx.x * 64 + wave * 16;    // row m = t*256 + b
  const int t = mbase >> 8, b0 = mbase & 255;       // 16 rows share one t
  const int nbase = blockIdx.y * 64;
  const unsigned short* arow = xbf + ((size_t)(b0 + n) * TT + t) * DD;
  f32x8 acc[4] = {};
  #pragma unroll
  for (int kt = 0; kt < 4; ++kt) {
    Frag16 a;
    const int kb = kt * 32 + half * 8;
    a.q[0] = *(const u32x4*)(arow + kb);
    a.q[1] = *(const u32x4*)(arow + kb + 16);
    #pragma unroll
    for (int jt = 0; jt < 4; ++jt) {
      const unsigned short* br = WcatT + (size_t)(nbase + jt * 16 + n) * DD + kt * 32 + half * 8;
      Frag16 b;
      b.q[0] = *(const u32x4*)(br);
      b.q[1] = *(const u32x4*)(br + 16);
      acc[jt] = WMMA_BF16(a.v, b.v, acc[jt]);
    }
  }
  #pragma unroll
  for (int jt = 0; jt < 4; ++jt) {
    const int col = nbase + jt * 16 + n;
    const float bias = Bcat[col];
    #pragma unroll
    for (int r = 0; r < 8; ++r) {
      const int m = mbase + half * 8 + r;
      XG[(size_t)m * GG + col] = f2bf(acc[jt][r] + bias);
    }
  }
}

// ---------------- kernel 4: backward GRU needs only its FIRST step (h0=0 on x[T-1]) ----------------
__global__ __launch_bounds__(256) void gru_b_single_kernel(
    const float* __restrict__ x, const float* __restrict__ Wi,
    const float* __restrict__ bi, const float* __restrict__ bh,
    float* __restrict__ hbF)
{
  const int id = blockIdx.x * 256 + threadIdx.x;    // 16384 = 256*64
  const int b = id >> 6, j = id & 63;
  const float* xr_ = x + ((size_t)b * TT + (TT - 1)) * DD;
  float s0 = bi[j], s1 = bi[64 + j], s2 = bi[128 + j];
  for (int d = 0; d < 128; ++d) {
    float xv = xr_[d];
    s0 += xv * Wi[(size_t)j * 128 + d];
    s1 += xv * Wi[(size_t)(64 + j) * 128 + d];
    s2 += xv * Wi[(size_t)(128 + j) * 128 + d];
  }
  float r = sigm(s0 + bh[j]);
  float z = sigm(s1 + bh[64 + j]);
  float nn = tanhf_(s2 + r * bh[128 + j]);
  hbF[id] = (1.f - z) * nn;                          // + z*0
}

// ---------------- kernel 5: the three recurrent scans, run concurrently ----------------
// role 0: GRU fwd (8 blocks), role 1: HB fwd (8), role 2: HB bwd (8)
__global__ __launch_bounds__(256) void scan_kernel(
    const unsigned short* __restrict__ XG,
    const unsigned short* __restrict__ WhB,
    const unsigned short* __restrict__ UTf,
    const unsigned short* __restrict__ UTb,
    const float* __restrict__ gf_bh,
    float* __restrict__ stable,
    float* __restrict__ hfF)
{
  __shared__ __align__(16) unsigned short sW[320 * 64];  // recurrent weights, bf16
  __shared__ __align__(16) unsigned short sH[32 * 64];   // h state, bf16 (A-matrix staging)
  const int role = blockIdx.x >> 3;
  const int bb   = (blockIdx.x & 7) * 32;
  const int tid  = threadIdx.x;
  const int wave = tid >> 5, lane = tid & 31;
  const int half = lane >> 4, n = lane & 15;
  const int mt = wave >> 2, c = wave & 3;                // wave = (M-tile, column group)
  const int j = c * 16 + n;                              // hidden-dim column owned by this lane
  const int rowb = bb + mt * 16 + half * 8;              // first of this lane's 8 batch rows

  const unsigned short* wsrc = (role == 0) ? WhB : (role == 1 ? UTf : UTb);
  const int wcount = (role == 0) ? 192 * 64 : 320 * 64;
#if HAVE_TDM
  // One TDM descriptor per block DMAs the whole recurrent weight matrix into LDS.
  if (wave == 0) {
    tdm_load_to_lds(wsrc, (unsigned)(uintptr_t)&sW[0], (unsigned)(wcount >> 1));
    __builtin_amdgcn_s_wait_tensorcnt(0);
  }
#else
  for (int i = tid; i < wcount; i += 256) sW[i] = wsrc[i];
#endif
  for (int i = tid; i < 32 * 64; i += 256) sH[i] = 0;
  __syncthreads();

  if (role == 0) {
    const float bhr = gf_bh[j], bhz = gf_bh[64 + j], bhn = gf_bh[128 + j];
    f32x8 h = {};
    for (int t = 0; t < TT; ++t) {
      Frag16 a0, a1;
      const unsigned short* hr_ = sH + (mt * 16 + n) * 64;
      a0.q[0] = *(const u32x4*)(hr_ + half * 8);
      a0.q[1] = *(const u32x4*)(hr_ + half * 8 + 16);
      a1.q[0] = *(const u32x4*)(hr_ + 32 + half * 8);
      a1.q[1] = *(const u32x4*)(hr_ + 32 + half * 8 + 16);
      f32x8 accg[3];
      #pragma unroll
      for (int g = 0; g < 3; ++g) {
        const unsigned short* br = sW + (size_t)((g * 4 + c) * 16 + n) * 64;
        Frag16 b0, b1;
        b0.q[0] = *(const u32x4*)(br + half * 8);
        b0.q[1] = *(const u32x4*)(br + half * 8 + 16);
        b1.q[0] = *(const u32x4*)(br + 32 + half * 8);
        b1.q[1] = *(const u32x4*)(br + 32 + half * 8 + 16);
        f32x8 acc = {};
        acc = WMMA_BF16(a0.v, b0.v, acc);
        acc = WMMA_BF16(a1.v, b1.v, acc);
        accg[g] = acc;
      }
      const unsigned short* xgp = XG + ((size_t)t * BB + rowb) * GG;
      #pragma unroll
      for (int r = 0; r < 8; ++r) {
        const unsigned short* xrow = xgp + (size_t)r * GG;
        float rr = sigm(bf2f(xrow[j])       + accg[0][r] + bhr);
        float zz = sigm(bf2f(xrow[64 + j])  + accg[1][r] + bhz);
        float nn = tanhf_(bf2f(xrow[128 + j]) + rr * (accg[2][r] + bhn));
        h[r] = (1.f - zz) * nn + zz * h[r];
      }
      __syncthreads();
      #pragma unroll
      for (int r = 0; r < 8; ++r)
        sH[(mt * 16 + half * 8 + r) * 64 + j] = f2bf(h[r]);
      __syncthreads();
      if (t < TT - 1)
        __builtin_prefetch((const void*)(XG + ((size_t)(t + 1) * BB + rowb) * GG + j), 0, 1);
    }
    #pragma unroll
    for (int r = 0; r < 8; ++r)
      hfF[(rowb + r) * 64 + j] = h[r];
  } else {
    const int colbase = (role == 1) ? 0 : 64;
    const int xoff    = (role == 1) ? 192 : 512;
    f32x8 kk = {}, kp = {};
    for (int s = 0; s < TT; ++s) {
      const int t = (role == 1) ? s : (TT - 1 - s);
      Frag16 a0, a1;
      const unsigned short* hr_ = sH + (mt * 16 + n) * 64;
      a0.q[0] = *(const u32x4*)(hr_ + half * 8);
      a0.q[1] = *(const u32x4*)(hr_ + half * 8 + 16);
      a1.q[0] = *(const u32x4*)(hr_ + 32 + half * 8);
      a1.q[1] = *(const u32x4*)(hr_ + 32 + half * 8 + 16);
      f32x8 accg[5];
      #pragma unroll
      for (int g = 0; g < 5; ++g) {                     // gates align in-lane across tiles c, c+4, ...
        const unsigned short* br = sW + (size_t)((g * 4 + c) * 16 + n) * 64;
        Frag16 b0, b1;
        b0.q[0] = *(const u32x4*)(br + half * 8);
        b0.q[1] = *(const u32x4*)(br + half * 8 + 16);
        b1.q[0] = *(const u32x4*)(br + 32 + half * 8);
        b1.q[1] = *(const u32x4*)(br + 32 + half * 8 + 16);
        f32x8 acc = {};
        acc = WMMA_BF16(a0.v, b0.v, acc);
        acc = WMMA_BF16(a1.v, b1.v, acc);
        accg[g] = acc;
      }
      const unsigned short* xgp = XG + ((size_t)t * BB + rowb) * GG + xoff;
      float hloc[8];
      #pragma unroll
      for (int r = 0; r < 8; ++r) {
        const unsigned short* xrow = xgp + (size_t)r * GG;
        float f_ = sigm(accg[0][r] + bf2f(xrow[j]));
        float i_ = sigm(accg[1][r] + bf2f(xrow[64 + j]));
        float o_ = sigm(accg[2][r] + bf2f(xrow[128 + j]));
        float u_ = sigm(accg[3][r] + bf2f(xrow[192 + j]));
        float kc = accg[4][r] + bf2f(xrow[256 + j]);
        kp[r] = u_ * tanhf_(kc) + (1.f - u_) * kp[r];
        kk[r] = f_ * kk[r] + i_ * kp[r];
        hloc[r] = o_ * tanhf_(kk[r]);
        stable[((size_t)(rowb + r) * TT + t) * 128 + colbase + j] = hloc[r];
      }
      __syncthreads();
      #pragma unroll
      for (int r = 0; r < 8; ++r)
        sH[(mt * 16 + half * 8 + r) * 64 + j] = f2bf(hloc[r]);
      __syncthreads();
    }
  }
}

// ---------------- kernel 6: attention hidden GEMM (WMMA) with fused tanh+logit epilogue ----------------
__global__ __launch_bounds__(128) void attn_kernel(
    const float* __restrict__ stable,
    const unsigned short* __restrict__ APT,
    const float* __restrict__ attnp_b,
    const float* __restrict__ attns_W,
    const float* __restrict__ attns_b,
    float* __restrict__ logits)
{
  const int wave = threadIdx.x >> 5, lane = threadIdx.x & 31;
  const int half = lane >> 4, n = lane & 15;
  const int mbase = blockIdx.x * 64 + wave * 16;     // row m = b*512 + t
  const float* arow = stable + (size_t)(mbase + n) * 128;
  f32x8 acc[4] = {};
  for (int kt = 0; kt < 4; ++kt) {
    const int kb = kt * 32 + half * 8;
    Frag16 a;
    #pragma unroll
    for (int i = 0; i < 8; ++i) a.h[i] = f2bf(arow[kb + i]);
    #pragma unroll
    for (int i = 0; i < 8; ++i) a.h[8 + i] = f2bf(arow[kb + 16 + i]);
    #pragma unroll
    for (int jt = 0; jt < 4; ++jt) {
      const unsigned short* br = APT + (size_t)(jt * 16 + n) * 128 + kt * 32 + half * 8;
      Frag16 b;
      b.q[0] = *(const u32x4*)(br);
      b.q[1] = *(const u32x4*)(br + 16);
      acc[jt] = WMMA_BF16(a.v, b.v, acc[jt]);
    }
  }
  float part[8];
  #pragma unroll
  for (int r = 0; r < 8; ++r) part[r] = 0.f;
  #pragma unroll
  for (int jt = 0; jt < 4; ++jt) {
    const int col = jt * 16 + n;
    const float wv = attns_W[col], bv = attnp_b[col];
    #pragma unroll
    for (int r = 0; r < 8; ++r) part[r] += tanhf_(acc[jt][r] + bv) * wv;
  }
  #pragma unroll
  for (int r = 0; r < 8; ++r) {
    part[r] += __shfl_xor(part[r], 8, 16);
    part[r] += __shfl_xor(part[r], 4, 16);
    part[r] += __shfl_xor(part[r], 2, 16);
    part[r] += __shfl_xor(part[r], 1, 16);
  }
  if (n == 0) {
    const float ab = attns_b[0];
    #pragma unroll
    for (int r = 0; r < 8; ++r)
      logits[mbase + half * 8 + r] = part[r] + ab;
  }
}

// ---------------- kernel 7: per-batch softmax over T + weighted context ----------------
__global__ __launch_bounds__(256) void softmax_ctx_kernel(
    const float* __restrict__ logits,
    const float* __restrict__ stable,
    float* __restrict__ ctxpre)
{
  __shared__ float sl[512];
  __shared__ float red[256];
  const int b = blockIdx.x, tid = threadIdx.x;
  float l0 = logits[b * TT + tid];
  float l1 = logits[b * TT + 256 + tid];
  red[tid] = fmaxf(l0, l1);
  __syncthreads();
  for (int o = 128; o > 0; o >>= 1) {
    if (tid < o) red[tid] = fmaxf(red[tid], red[tid + o]);
    __syncthreads();
  }
  const float mx = red[0];
  __syncthreads();
  float e0 = __expf(l0 - mx), e1 = __expf(l1 - mx);
  sl[tid] = e0; sl[256 + tid] = e1;
  red[tid] = e0 + e1;
  __syncthreads();
  for (int o = 128; o > 0; o >>= 1) {
    if (tid < o) red[tid] += red[tid + o];
    __syncthreads();
  }
  const float inv = 1.f / red[0];
  __syncthreads();
  const int d = tid & 127, hf = tid >> 7;
  const float* sp = stable + ((size_t)b * TT + hf * 256) * 128 + d;
  float acc = 0.f;
  for (int t2 = 0; t2 < 256; ++t2) acc += sl[hf * 256 + t2] * sp[(size_t)t2 * 128];
  red[tid] = acc;
  __syncthreads();
  if (tid < 128) ctxpre[b * 128 + tid] = (red[tid] + red[tid + 128]) * inv;
}

// ---------------- kernel 8: all small head MLPs (one block per batch row) ----------------
__global__ __launch_bounds__(128) void final_kernel(
    const float* __restrict__ hfF, const float* __restrict__ hbF,
    const float* __restrict__ ctxpre,
    const float* __restrict__ fast_W, const float* __restrict__ fast_b,
    const float* __restrict__ stab_W, const float* __restrict__ stab_b,
    const float* __restrict__ gW1, const float* __restrict__ gb1,
    const float* __restrict__ gW2, const float* __restrict__ gb2,
    const float* __restrict__ hW1, const float* __restrict__ hb1,
    const float* __restrict__ hW2, const float* __restrict__ hb2,
    float* __restrict__ out)
{
  __shared__ float s_in[128], s_fl[128], s_ctx[128], s_gh[128], s_fu[128], s_red[128];
  const int b = blockIdx.x, tid = threadIdx.x;
  s_in[tid] = (tid < 64) ? hfF[b * 64 + tid] : hbF[b * 64 + tid - 64];
  __syncthreads();
  float acc = fast_b[tid];
  for (int k = 0; k < 128; ++k) acc += s_in[k] * fast_W[k * 128 + tid];
  s_fl[tid] = acc;                                   // fast_last
  __syncthreads();
  s_in[tid] = ctxpre[b * 128 + tid];
  __syncthreads();
  acc = stab_b[tid];
  for (int k = 0; k < 128; ++k) acc += s_in[k] * stab_W[k * 128 + tid];
  s_ctx[tid] = acc;                                  // stable_ctx
  __syncthreads();
  acc = gb1[tid];
  for (int k = 0; k < 128; ++k)
    acc += s_fl[k] * gW1[k * 128 + tid] + s_ctx[k] * gW1[(128 + k) * 128 + tid];
  s_gh[tid] = fmaxf(acc, 0.f);
  __syncthreads();
  s_red[tid] = s_gh[tid] * gW2[tid];
  __syncthreads();
  for (int o = 64; o > 0; o >>= 1) {
    if (tid < o) s_red[tid] += s_red[tid + o];
    __syncthreads();
  }
  const float gate = sigm(s_red[0] + gb2[0]);
  __syncthreads();
  s_fu[tid] = gate * s_fl[tid] + (1.f - gate) * s_ctx[tid];
  __syncthreads();
  acc = hb1[tid];
  for (int k = 0; k < 128; ++k)
    acc += s_fu[k] * hW1[k * 128 + tid]
         + s_fl[k] * hW1[(128 + k) * 128 + tid]
         + s_ctx[k] * hW1[(256 + k) * 128 + tid];
  const float h1 = fmaxf(acc, 0.f);
  s_red[tid] = h1 * hW2[tid];
  __syncthreads();
  for (int o = 64; o > 0; o >>= 1) {
    if (tid < o) s_red[tid] += s_red[tid + o];
    __syncthreads();
  }
  if (tid == 0) out[b] = s_red[0] + hb2[0];
}

// ---------------- launcher ----------------
extern "C" void kernel_launch(void* const* d_in, const int* in_sizes, int n_in,
                              void* d_out, int out_size, void* d_ws, size_t ws_size,
                              hipStream_t stream)
{
  (void)in_sizes; (void)n_in; (void)out_size; (void)ws_size;
  const float* x       = (const float*)d_in[0];
  const float* gf_Wi   = (const float*)d_in[1];
  const float* gf_Wh   = (const float*)d_in[2];
  const float* gf_bi   = (const float*)d_in[3];
  const float* gf_bh   = (const float*)d_in[4];
  const float* gb_Wi   = (const float*)d_in[5];
  const float* gb_bi   = (const float*)d_in[7];
  const float* gb_bh   = (const float*)d_in[8];
  const float* hf_W    = (const float*)d_in[9];
  const float* hf_b    = (const float*)d_in[10];
  const float* hf_U    = (const float*)d_in[11];
  const float* hf_c    = (const float*)d_in[12];
  const float* hb_W    = (const float*)d_in[13];
  const float* hb_b    = (const float*)d_in[14];
  const float* hb_U    = (const float*)d_in[15];
  const float* hb_c    = (const float*)d_in[16];
  const float* fast_W  = (const float*)d_in[17];
  const float* fast_b  = (const float*)d_in[18];
  const float* stab_W  = (const float*)d_in[19];
  const float* stab_b  = (const float*)d_in[20];
  const float* attnp_W = (const float*)d_in[21];
  const float* attnp_b = (const float*)d_in[22];
  const float* attns_W = (const float*)d_in[23];
  const float* attns_b = (const float*)d_in[24];
  const float* gate_W1 = (const float*)d_in[25];
  const float* gate_b1 = (const float*)d_in[26];
  const float* gate_W2 = (const float*)d_in[27];
  const float* gate_b2 = (const float*)d_in[28];
  const float* head_W1 = (const float*)d_in[29];
  const float* head_b1 = (const float*)d_in[30];
  const float* head_W2 = (const float*)d_in[31];
  const float* head_b2 = (const float*)d_in[32];

  char* base = (char*)d_ws;
  size_t off = 0;
  auto carve = [&](size_t bytes) -> void* {
    void* p = base + off;
    off = (off + bytes + 255) & ~(size_t)255;
    return p;
  };
  unsigned short* XG    = (unsigned short*)carve((size_t)TT * BB * GG * 2);   // 218 MB bf16
  unsigned short* xbf   = (unsigned short*)carve((size_t)BB * TT * DD * 2);   // 33.5 MB
  float*          stab  = (float*)carve((size_t)BB * TT * 128 * 4);           // 67 MB
  unsigned short* WcatT = (unsigned short*)carve((size_t)GG * DD * 2);
  float*          Bcat  = (float*)carve((size_t)GG * 4);
  unsigned short* WhB   = (unsigned short*)carve((size_t)192 * 64 * 2);
  unsigned short* UTf   = (unsigned short*)carve((size_t)320 * 64 * 2);
  unsigned short* UTb   = (unsigned short*)carve((size_t)320 * 64 * 2);
  unsigned short* APT   = (unsigned short*)carve((size_t)64 * 128 * 2);
  float*          logz  = (float*)carve((size_t)BB * TT * 4);
  float*          hfF   = (float*)carve((size_t)BB * 64 * 4);
  float*          hbF   = (float*)carve((size_t)BB * 64 * 4);
  float*          ctxp  = (float*)carve((size_t)BB * 128 * 4);

  pack_kernel<<<416, 256, 0, stream>>>(gf_Wi, gf_bi, gf_Wh, hf_W, hf_b, hf_U, hf_c,
                                       hb_W, hb_b, hb_U, hb_c, attnp_W,
                                       WcatT, Bcat, WhB, UTf, UTb, APT);
  cvt_x_kernel<<<16384, 256, 0, stream>>>(x, xbf);
  gemm_xg_kernel<<<dim3(2048, 13), 128, 0, stream>>>(xbf, WcatT, Bcat, XG);
  gru_b_single_kernel<<<64, 256, 0, stream>>>(x, gb_Wi, gb_bi, gb_bh, hbF);
  scan_kernel<<<24, 256, 0, stream>>>(XG, WhB, UTf, UTb, gf_bh, stab, hfF);
  attn_kernel<<<2048, 128, 0, stream>>>(stab, APT, attnp_b, attns_W, attns_b, logz);
  softmax_ctx_kernel<<<256, 256, 0, stream>>>(logz, stab, ctxp);
  final_kernel<<<256, 128, 0, stream>>>(hfF, hbF, ctxp, fast_W, fast_b, stab_W, stab_b,
                                        gate_W1, gate_b1, gate_W2, gate_b2,
                                        head_W1, head_b1, head_W2, head_b2,
                                        (float*)d_out);
}
